// WHVIStackedMatrix_54348516164120
// MI455X (gfx1250) — matrix operand order repaired and checked by reading, not compiled, not run on hardware
//
#include <hip/hip_runtime.h>
#include <hip/hip_bf16.h>

typedef __attribute__((ext_vector_type(16))) _Float16 v16h;
typedef __attribute__((ext_vector_type(8)))  float    v8f;
typedef __attribute__((ext_vector_type(8)))  float    float8;

#define D_IN   256
#define STACK  4
#define D_OUT  1024
#define BATCH  65536

// ---------------------------------------------------------------------------
// Phase 1: build W in f16.
//   g_eff = g_mu + softplus(g_rho) * eps           (square_term is linear in g)
//   (H diag(g) H)[i][j] == FWHT(g)[i ^ j]          (Sylvester Hadamard identity)
//   W[st*256 + i][j] = s1[st][i] * s2[st][j] * FWHT(g_eff_st)[i ^ j]
// One block per output row; each block redundantly does the tiny 256-pt FWHT
// of its stack's g_eff in LDS (trivial cost vs. the 34-GFLOP GEMM).
// ---------------------------------------------------------------------------
__global__ void whvi_build_w(const float* __restrict__ s1,
                             const float* __restrict__ s2,
                             const float* __restrict__ g_mu,
                             const float* __restrict__ g_rho,
                             const float* __restrict__ eps,
                             _Float16* __restrict__ Wh) {
  __shared__ float h[D_IN];
  const int st   = blockIdx.x >> 8;   // stack index 0..3
  const int i    = blockIdx.x & 255;  // row-in-stack 0..255
  const int j    = threadIdx.x;       // column 0..255
  const int base = st * D_IN;

  const float rho  = g_rho[base + j];
  const float geff = g_mu[base + j] + log1pf(__expf(rho)) * eps[base + j];
  h[j] = geff;
  __syncthreads();

  // Unnormalized fast Walsh-Hadamard transform (== H @ g_eff for Sylvester H)
  for (int s = 1; s < D_IN; s <<= 1) {
    const float a = h[j];
    const float b = h[j ^ s];
    __syncthreads();
    h[j] = (j & s) ? (b - a) : (a + b);
    __syncthreads();
  }

  const float w = s1[base + i] * s2[base + j] * h[i ^ j];
  Wh[(size_t)(base + i) * D_IN + j] = (_Float16)w;
}

// ---------------------------------------------------------------------------
// Phase 2: out = X @ W^T  (M=65536, N=1024, K=256) with f16 WMMA, f32 accum.
// One wave computes a 16(M) x 64(N) tile: one A fragment feeds 4 B fragments
// (4 accumulators) per K=32 step, 8 K-steps total.
//
// A (16x32 f16) lane layout (contiguous k, as B): lanes 0-15 hold M=lane, K=0..15;
// lanes 16-31 hold M=lane-16, K=16..31. Built from two 32B f32 loads
// + v_cvt per lane.
// B (32x16 f16): lane holds column N=lane%16; lanes 0-15 K=0..15,
// lanes 16-31 K=16..31 -> one contiguous 32B load from a row-major W row.
// C/D (16x16 f32): VGPR r, lane L -> row r + 8*(L>=16), col L%16.
// ---------------------------------------------------------------------------
__global__ void whvi_gemm(const float* __restrict__ X,
                          const _Float16* __restrict__ Wh,
                          float* __restrict__ Out) {
  const int lane = threadIdx.x & 31;
  const int wave = threadIdx.x >> 5;
  const int tile = blockIdx.x * 8 + wave;   // 65536 wave-tiles total
  const int mTile = tile >> 4;              // / (1024/64)
  const int nTile = tile & 15;
  const int m0 = mTile << 4;                // * 16
  const int n0 = nTile << 6;                // * 64
  const int l15 = lane & 15;
  const int hs  = lane >> 4;                // lane-half select

  const float*    xrow = X  + (size_t)(m0 + l15) * D_IN;
  const _Float16* wrow = Wh + (size_t)(n0 + l15) * D_IN + hs * 16;

  v8f acc0 = {}, acc1 = {}, acc2 = {}, acc3 = {};

#pragma unroll
  for (int kb = 0; kb < D_IN; kb += 32) {
    // A fragment: K group0 = kb + 16*hs + [0..7], group1 = kb + 16*hs + 8 + [0..7]
    const float8 g0 = *(const float8*)(xrow + kb + hs * 16);
    const float8 g1 = *(const float8*)(xrow + kb + hs * 16 + 8);
    v16h a;
#pragma unroll
    for (int e = 0; e < 8; ++e) {
      a[e]     = (_Float16)g0[e];
      a[e + 8] = (_Float16)g1[e];
    }

    // B fragments: 32B contiguous per lane from W rows n0+t*16+l15
    const v16h b0 = *(const v16h*)(wrow + kb);
    const v16h b1 = *(const v16h*)(wrow + 16 * D_IN + kb);
    const v16h b2 = *(const v16h*)(wrow + 32 * D_IN + kb);
    const v16h b3 = *(const v16h*)(wrow + 48 * D_IN + kb);

    acc0 = __builtin_amdgcn_wmma_f32_16x16x32_f16(false, a, false, b0, (short)0, acc0, false, false);
    acc1 = __builtin_amdgcn_wmma_f32_16x16x32_f16(false, a, false, b1, (short)0, acc1, false, false);
    acc2 = __builtin_amdgcn_wmma_f32_16x16x32_f16(false, a, false, b2, (short)0, acc2, false, false);
    acc3 = __builtin_amdgcn_wmma_f32_16x16x32_f16(false, a, false, b3, (short)0, acc3, false, false);
  }

  // Store: row = m0 + 8*hs + r, cols n0 + t*16 + l15
  float* obase = Out + (size_t)(m0 + hs * 8) * D_OUT + n0 + l15;
#pragma unroll
  for (int r = 0; r < 8; ++r) {
    float* orow = obase + (size_t)r * D_OUT;
    orow[0]  = acc0[r];
    orow[16] = acc1[r];
    orow[32] = acc2[r];
    orow[48] = acc3[r];
  }
}

extern "C" void kernel_launch(void* const* d_in, const int* in_sizes, int n_in,
                              void* d_out, int out_size, void* d_ws, size_t ws_size,
                              hipStream_t stream) {
  (void)in_sizes; (void)n_in; (void)out_size; (void)ws_size;
  const float* x     = (const float*)d_in[0];
  const float* s1    = (const float*)d_in[1];
  const float* s2    = (const float*)d_in[2];
  const float* g_mu  = (const float*)d_in[3];
  const float* g_rho = (const float*)d_in[4];
  const float* eps   = (const float*)d_in[5];
  _Float16* Wh = (_Float16*)d_ws;           // 1024 x 256 f16 = 512 KB scratch
  float* out = (float*)d_out;

  // Build W (tiny), then the memory-bound WMMA GEMM.
  whvi_build_w<<<D_OUT, D_IN, 0, stream>>>(s1, s2, g_mu, g_rho, eps, Wh);

  const int waveTiles = (BATCH / 16) * (D_OUT / 64);  // 65536
  whvi_gemm<<<waveTiles / 8, 256, 0, stream>>>(x, Wh, out);
}
